// ProteinGN_10419590660562
// MI455X (gfx1250) — compile-verified, matrix operand order checked by reading, not run on hardware
//
#include <hip/hip_runtime.h>

typedef __attribute__((ext_vector_type(16))) _Float16 v16h;
typedef __attribute__((ext_vector_type(8)))  _Float16 v8h;
typedef __attribute__((ext_vector_type(8)))  float    v8f;

#define LDS_FENCE() asm volatile("s_wait_dscnt 0" ::: "memory")

// CDNA5 async global->LDS copy path (ASYNCcnt-tracked). Set to 0 to fall back
// to the plain load + ds_store path.
#define ASYNC_GATHER 1

__device__ __forceinline__ void gather16(const _Float16* __restrict__ gsrc,
                                         _Float16* lds_dst) {
#if ASYNC_GATHER
  unsigned ldsa = (unsigned)(size_t)lds_dst;  // low 32 bits = LDS byte offset
  asm volatile("global_load_async_to_lds_b128 %0, %1, off"
               :: "v"(ldsa), "v"(gsrc)
               : "memory");
#else
  *(v8h*)lds_dst = *(const v8h*)gsrc;
#endif
}

#if ASYNC_GATHER
#define GATHER_WAIT() asm volatile("s_wait_asynccnt 0x0" ::: "memory")
#else
#define GATHER_WAIT() LDS_FENCE()
#endif

constexpr int NN  = 65536;    // nodes
constexpr int NE  = 1572864;  // edges
constexpr int NGR = 256;      // graphs
constexpr int EPG = 6144;     // edges per graph
constexpr int NPG = 256;      // nodes per graph
constexpr int WPB = 4;        // waves per block for plain GEMM kernels

static inline int cdiv(long a, long b) { return (int)((a + b - 1) / b); }
constexpr int ceil16(int x) { return (x + 15) & ~15; }
constexpr int ceil32(int x) { return (x + 31) & ~31; }

// ---------------------------------------------------------------------------
// WMMA 16-bit operand loaders (v_wmma_f32_16x16x32_f16 layout):
// lane L<16 holds row L, halves K = {kb+0..7, kb+16..23};
// lane L>=16 holds row L-16, halves K = {kb+8..15, kb+24..31}.
// A rows: width kdim halves (multiple of 8, rows >=16B aligned); chunk
// validity is whole-chunk & compile-time foldable -> unconditional b128 loads
// (+ rare select; masked case may over-read 16B inside the oversized carve).
// B rows: padded weights (stride multiple of 32 halves) -> fully unconditional.
// ---------------------------------------------------------------------------
__device__ __forceinline__ v16h frag_cat(v8h c0, v8h c1) {
  return __builtin_shufflevector(c0, c1, 0, 1, 2, 3, 4, 5, 6, 7,
                                 8, 9, 10, 11, 12, 13, 14, 15);
}

__device__ __forceinline__ v16h load_frag(const _Float16* __restrict__ row,
                                          int kdim, int kbase, int hf) {
  v8h z = {};
  v8h c0 = z, c1 = z;
  if (kbase + 16 <= kdim) {
    c0 = *(const v8h*)(row + kbase + hf * 8);
  } else if (kbase + 8 <= kdim) {
    v8h t = *(const v8h*)(row + kbase + hf * 8);  // benign over-read for hf=1
    if (hf == 0) c0 = t;
  }
  if (kbase + 32 <= kdim) {
    c1 = *(const v8h*)(row + kbase + 16 + hf * 8);
  } else if (kbase + 24 <= kdim) {
    v8h t = *(const v8h*)(row + kbase + 16 + hf * 8);
    if (hf == 0) c1 = t;
  }
  return frag_cat(c0, c1);
}

__device__ __forceinline__ v16h load_frag_full(const _Float16* __restrict__ row,
                                               int kbase, int hf) {
  v8h c0 = *(const v8h*)(row + kbase + hf * 8);
  v8h c1 = *(const v8h*)(row + kbase + 16 + hf * 8);
  return frag_cat(c0, c1);
}

// ---------------------------------------------------------------------------
// Generic GEMM: out[M,NP] = act(A[M,KD] @ B + bias).
// B pre-transposed & padded: Bt[NP][KP], KP = ceil32(KD), zero-filled.
// ---------------------------------------------------------------------------
template<int KD, int NP, bool RELU>
__global__ void wmma_gemm(const _Float16* __restrict__ A, const _Float16* __restrict__ Bt,
                          const float* __restrict__ bias, _Float16* __restrict__ out, int M) {
  int wave = blockIdx.x * (blockDim.x >> 5) + (threadIdx.x >> 5);
  int lane = threadIdx.x & 31;
  int row0 = wave * 16;
  if (row0 >= M) return;
  int r = lane & 15, hf = lane >> 4;
  constexpr int KP = ceil32(KD);
  constexpr int KC = KP / 32;
  constexpr int NT = NP / 16;
  const _Float16* arow = A + (size_t)(row0 + r) * KD;
  v16h a[KC];
#pragma unroll
  for (int kc = 0; kc < KC; ++kc) a[kc] = load_frag(arow, KD, kc * 32, hf);
#pragma unroll
  for (int nt = 0; nt < NT; ++nt) {
    v8f c = {};
    const _Float16* brow = Bt + (size_t)(nt * 16 + r) * KP;
#pragma unroll
    for (int kc = 0; kc < KC; ++kc) {
      v16h b = load_frag_full(brow, kc * 32, hf);
      c = __builtin_amdgcn_wmma_f32_16x16x32_f16(false, a[kc], false, b,
                                                 (short)0, c, false, false);
    }
    int ncol = nt * 16 + r;
    float bv = bias ? bias[ncol] : 0.0f;
#pragma unroll
    for (int i = 0; i < 8; ++i) {
      int m = hf * 8 + i;
      float v = c[i] + bv;
      if (RELU) v = fmaxf(v, 0.0f);
      out[(size_t)(row0 + m) * NP + ncol] = (_Float16)v;
    }
  }
}

// ---------------------------------------------------------------------------
// Edge update, one workgroup per (graph, column-slab):
//   e_new = relu(e@We + ns[senders] + ge[graph] + be)
// All segment sums stay in LDS:
//   - incoming-edge sums accumulated with ds_add_f32 into inc_acc[256][COLS],
//     flushed at the end as f16 means (x 1/indeg) with plain coalesced stores
//     -> zero global atomics, no global memset, no f32 round-trip.
//   - per-graph edge sum in eg_acc[COLS], plain store at the end.
// Sender rows are gathered straight into LDS with async global->LDS loads
// (ASYNCcnt); the staging tile is then reused in-place for the output tile so
// e_out is written back as coalesced b128 stores.
// 16 waves (512 threads) per workgroup; each wave walks 16-edge tiles.
// ---------------------------------------------------------------------------
template<int IE, int OE, int COLS>
__global__ void edge_update(const _Float16* __restrict__ e_in, const _Float16* __restrict__ Wet,
                            const _Float16* __restrict__ ns, const float* __restrict__ ge,
                            const float* __restrict__ be,
                            const int* __restrict__ senders, const int* __restrict__ receivers,
                            const int* __restrict__ indeg,
                            _Float16* __restrict__ e_out, _Float16* __restrict__ inch,
                            float* __restrict__ e_gsum) {
  constexpr int NW = 16;          // waves per workgroup
  constexpr int SLABS = OE / COLS;
  constexpr int CPR = COLS / 8;   // 16B chunks per staged row
  __shared__ float inc_acc[NPG * COLS];
  __shared__ float eg_acc[COLS];
  __shared__ __align__(16) _Float16 stage[NW][16 * COLS];
  __shared__ int sidx[NW][16];
  __shared__ int ridx[NW][16];
  __shared__ float red[NW][32];

  int tid = threadIdx.x;
  int w = tid >> 5;
  int lane = tid & 31;
  int graph = blockIdx.x / SLABS;
  int c0 = (blockIdx.x % SLABS) * COLS;
  int r = lane & 15, hf = lane >> 4;

  for (int i = tid; i < NPG * COLS; i += NW * 32) inc_acc[i] = 0.0f;
  for (int i = tid; i < COLS; i += NW * 32) eg_acc[i] = 0.0f;
  __syncthreads();

  constexpr int KP = ceil32(IE);
  constexpr int KC = KP / 32;
  constexpr int TILES = EPG / 16;  // 384

  for (int t = w; t < TILES; t += NW) {
    int bse = graph * EPG + t * 16;
    if (lane < 16) {
      sidx[w][lane] = senders[bse + lane];
      ridx[w][lane] = receivers[bse + lane];
    }
    LDS_FENCE();  // also drains last iteration's stage reads before overwrite
    // async gather of the 16 sender rows (our column slab) into LDS
    for (int i = lane; i < 16 * CPR; i += 32) {
      int row = i / CPR, off = i % CPR;
      gather16(ns + (size_t)sidx[w][row] * OE + c0 + off * 8, &stage[w][i * 8]);
    }
    const _Float16* arow = e_in + (size_t)(bse + r) * IE;
    if (t + NW < TILES) __builtin_prefetch(arow + (size_t)NW * 16 * IE, 0, 1);
    v16h a[KC];
#pragma unroll
    for (int kc = 0; kc < KC; ++kc) a[kc] = load_frag(arow, IE, kc * 32, hf);
    int rec[8];
#pragma unroll
    for (int i = 0; i < 8; ++i) rec[i] = ridx[w][hf * 8 + i] & (NPG - 1);
    GATHER_WAIT();
#pragma unroll
    for (int nt = 0; nt < COLS / 16; ++nt) {
      v8f c = {};
      const _Float16* brow = Wet + (size_t)(c0 + nt * 16 + r) * KP;
#pragma unroll
      for (int kc = 0; kc < KC; ++kc) {
        v16h b = load_frag_full(brow, kc * 32, hf);
        c = __builtin_amdgcn_wmma_f32_16x16x32_f16(false, a[kc], false, b,
                                                   (short)0, c, false, false);
      }
      int cl = nt * 16 + r;            // column within slab
      int cg = c0 + cl;                // global output column
      float bv = be[cg] + ge[graph * OE + cg];
      float part = 0.0f;
#pragma unroll
      for (int i = 0; i < 8; ++i) {
        int m = hf * 8 + i;
        float v = c[i] + (float)stage[w][m * COLS + cl] + bv;
        v = fmaxf(v, 0.0f);
        stage[w][m * COLS + cl] = (_Float16)v;   // in-place: becomes out tile
        atomicAdd(&inc_acc[rec[i] * COLS + cl], v);  // LDS atomic (ds_add_f32)
        part += v;
      }
      red[w][lane] = part;
      LDS_FENCE();
      if (hf == 0) atomicAdd(&eg_acc[cl], part + red[w][lane + 16]);
    }
    LDS_FENCE();
    // coalesced b128 store of the finished 16 x COLS output tile
    {
      const v8h* src = (const v8h*)&stage[w][0];
      for (int i = lane; i < 16 * CPR; i += 32) {
        int row = i / CPR, off = i % CPR;
        *(v8h*)(e_out + (size_t)(bse + row) * OE + c0 + off * 8) = src[i];
      }
    }
  }
  __syncthreads();
  // flush incoming means (f16, already scaled by 1/indeg) and per-graph sum
  for (int i = tid; i < NPG * COLS; i += NW * 32) {
    int nloc = i / COLS, c = i % COLS;
    int node = (graph << 8) + nloc;
    int d = indeg[node];
    float dn = (float)(d > 0 ? d : 1);
    inch[(size_t)node * OE + c0 + c] = (_Float16)(inc_acc[i] / dn);
  }
  for (int c = tid; c < COLS; c += NW * 32)
    e_gsum[graph * OE + c0 + c] = eg_acc[c];
}

// ---------------------------------------------------------------------------
// Node update (fused): n_new = relu(n@Wn + inc_mean@Wi + gn[graph] + bn),
// plus per-graph node sum accumulation.
// ---------------------------------------------------------------------------
template<int IN_, int OE, int ON>
__global__ void node_update(const _Float16* __restrict__ n_in, const _Float16* __restrict__ Wnt,
                            const _Float16* __restrict__ inc_h, const _Float16* __restrict__ Wit,
                            const float* __restrict__ gn, const float* __restrict__ bn,
                            _Float16* __restrict__ n_out, float* __restrict__ n_gsum, int nnodes) {
  __shared__ float red[WPB][32];
  int w = threadIdx.x >> 5;
  int lane = threadIdx.x & 31;
  int tile = blockIdx.x * WPB + w;
  int bse = tile * 16;
  if (bse >= nnodes) return;
  int r = lane & 15, hf = lane >> 4;
  int graph = bse / NPG;
  constexpr int KP1 = ceil32(IN_);
  constexpr int KP2 = ceil32(OE);
  constexpr int KC1 = KP1 / 32;
  constexpr int KC2 = KP2 / 32;
  v16h a1[KC1], a2[KC2];
  const _Float16* nrow = n_in + (size_t)(bse + r) * IN_;
  const _Float16* irow = inc_h + (size_t)(bse + r) * OE;
#pragma unroll
  for (int kc = 0; kc < KC1; ++kc) a1[kc] = load_frag(nrow, IN_, kc * 32, hf);
#pragma unroll
  for (int kc = 0; kc < KC2; ++kc) a2[kc] = load_frag(irow, OE, kc * 32, hf);
#pragma unroll
  for (int nt = 0; nt < ON / 16; ++nt) {
    v8f c = {};
    const _Float16* brow1 = Wnt + (size_t)(nt * 16 + r) * KP1;
    const _Float16* brow2 = Wit + (size_t)(nt * 16 + r) * KP2;
#pragma unroll
    for (int kc = 0; kc < KC1; ++kc) {
      v16h b = load_frag_full(brow1, kc * 32, hf);
      c = __builtin_amdgcn_wmma_f32_16x16x32_f16(false, a1[kc], false, b,
                                                 (short)0, c, false, false);
    }
#pragma unroll
    for (int kc = 0; kc < KC2; ++kc) {
      v16h b = load_frag_full(brow2, kc * 32, hf);
      c = __builtin_amdgcn_wmma_f32_16x16x32_f16(false, a2[kc], false, b,
                                                 (short)0, c, false, false);
    }
    int ncol = nt * 16 + r;
    float bv = bn[ncol] + gn[graph * ON + ncol];
    float part = 0.0f;
#pragma unroll
    for (int i = 0; i < 8; ++i) {
      int m = hf * 8 + i;
      float v = fmaxf(c[i] + bv, 0.0f);
      n_out[(size_t)(bse + m) * ON + ncol] = (_Float16)v;
      part += v;
    }
    red[w][lane] = part;
    LDS_FENCE();
    if (hf == 0) atomicAdd(&n_gsum[graph * ON + ncol], part + red[w][lane + 16]);
  }
}

// ------------------------- small scalar kernels ---------------------------

// W[fi,fo] f32 -> Wt[NP][KP] f16 transposed, zero padded
__global__ void w_to_f16t_pad(const float* __restrict__ W, _Float16* __restrict__ Wt,
                              int fi, int fo, int KP, int NP) {
  int i = blockIdx.x * blockDim.x + threadIdx.x;
  if (i >= NP * KP) return;
  int c = i / KP, k = i % KP;
  Wt[i] = (k < fi && c < fo) ? (_Float16)W[(size_t)k * fo + c] : (_Float16)0.0f;
}

__global__ void pad_bias(const float* __restrict__ b, float* __restrict__ out, int n, int np) {
  int i = blockIdx.x * blockDim.x + threadIdx.x;
  if (i < np) out[i] = (i < n) ? b[i] : 0.0f;
}

// node features f32 [N,83] -> f16 [N,96] zero padded
__global__ void node_to_f16_pad(const float* __restrict__ x, _Float16* __restrict__ y) {
  long i = (long)blockIdx.x * blockDim.x + threadIdx.x;
  if (i >= (long)NN * 96) return;
  int rI = (int)(i / 96), c = (int)(i % 96);
  y[i] = (c < 83) ? (_Float16)x[(size_t)rI * 83 + c] : (_Float16)0.0f;
}

__global__ void edge_encoder(const float* __restrict__ ef, const float* __restrict__ W1,
                             const float* __restrict__ b1, const float* __restrict__ W2,
                             const float* __restrict__ b2, _Float16* __restrict__ out, int ne) {
  int e = blockIdx.x * blockDim.x + threadIdx.x;
  if (e >= ne) return;
  float x0 = ef[(size_t)e * 2 + 0], x1 = ef[(size_t)e * 2 + 1];
  float h[4];
#pragma unroll
  for (int c = 0; c < 4; ++c)
    h[c] = fmaxf(x0 * W1[0 * 4 + c] + x1 * W1[1 * 4 + c] + b1[c], 0.0f);
#pragma unroll
  for (int c = 0; c < 8; ++c) {
    float a = b2[c];
#pragma unroll
    for (int k = 0; k < 4; ++k) a += h[k] * W2[k * 8 + c];
    out[(size_t)e * 8 + c] = (_Float16)fmaxf(a, 0.0f);
  }
}

__global__ void glob_encoder(const float* __restrict__ gf, const float* __restrict__ W,
                             const float* __restrict__ b, float* __restrict__ out) {
  int i = blockIdx.x * blockDim.x + threadIdx.x;
  if (i >= NGR * 4) return;
  int g = i / 4, c = i % 4;
  out[i] = fmaxf(gf[g] * W[c] + b[c], 0.0f);
}

__global__ void small_mm(const float* __restrict__ A, const float* __restrict__ W,
                         float* __restrict__ out, int R, int K, int Nd) {
  int i = blockIdx.x * blockDim.x + threadIdx.x;
  if (i >= R * Nd) return;
  int rI = i / Nd, c = i % Nd;
  float a = 0.0f;
  for (int k = 0; k < K; ++k) a += A[(size_t)rI * K + k] * W[(size_t)k * Nd + c];
  out[i] = a;
}

__global__ void count_indeg(const int* __restrict__ recv, int* __restrict__ indeg, int ne) {
  int i = blockIdx.x * blockDim.x + threadIdx.x;
  if (i < ne) atomicAdd(&indeg[recv[i]], 1);
}

__global__ void glob_update(const float* __restrict__ e_gsum, const float* __restrict__ n_gsum,
                            const float* __restrict__ g, const float* __restrict__ We2,
                            const float* __restrict__ Wn2, const float* __restrict__ Wg3,
                            const float* __restrict__ bg, int OE, int ON, int IG, int OG,
                            float einv, float ninv, float* __restrict__ g_out) {
  int i = blockIdx.x * blockDim.x + threadIdx.x;
  if (i >= NGR * OG) return;
  int gi = i / OG, c = i % OG;
  float acc = bg[c];
  for (int k = 0; k < OE; ++k) acc += e_gsum[(size_t)gi * OE + k] * einv * We2[(size_t)k * OG + c];
  for (int k = 0; k < ON; ++k) acc += n_gsum[(size_t)gi * ON + k] * ninv * Wn2[(size_t)k * OG + c];
  for (int k = 0; k < IG; ++k) acc += g[(size_t)gi * IG + k] * Wg3[(size_t)k * OG + c];
  g_out[i] = fmaxf(acc, 0.0f);
}

__global__ void node_readout(const _Float16* __restrict__ n, const float* __restrict__ W,
                             const float* __restrict__ b, float* __restrict__ out,
                             float* __restrict__ nsum) {
  int i = blockIdx.x * blockDim.x + threadIdx.x;
  if (i >= NN) return;
  float a = b[0];
#pragma unroll
  for (int k8 = 0; k8 < 16; ++k8) {
    v8h v = *(const v8h*)(n + (size_t)i * 128 + k8 * 8);
#pragma unroll
    for (int j = 0; j < 8; ++j) a += (float)v[j] * W[k8 * 8 + j];
  }
  float s = 1.0f / (1.0f + __expf(-a));
  out[i] = s;
  atomicAdd(&nsum[i >> 8], s);
}

__global__ void glob_readout(const float* __restrict__ g, const float* __restrict__ Wg,
                             const float* __restrict__ Wn, const float* __restrict__ br,
                             const float* __restrict__ nsum, float* __restrict__ out) {
  int gi = blockIdx.x * blockDim.x + threadIdx.x;
  if (gi >= NGR) return;
  float a = br[0] + (nsum[gi] * (1.0f / NPG)) * Wn[0];
  for (int k = 0; k < 32; ++k) a += g[(size_t)gi * 32 + k] * Wg[k];
  out[gi] = 1.0f / (1.0f + __expf(-a));
}

// ----------------------------- host-side layer ----------------------------

struct LayerPtrs {
  const _Float16 *e_in, *n_in; const float* g_in;
  _Float16 *e_out, *n_out; float* g_out;
  const _Float16 *Wet, *Wst, *Wnt, *Wit;
  const float *Wg_e, *be, *Wg2_n, *bn, *We2, *Wn2, *Wg3, *bg;
  const int *senders, *receivers, *indeg;
  _Float16 *ns_buf, *inc_h;
  float *ge, *gn, *e_gsum, *n_gsum;
};

template<int IE, int IN_, int IG, int OE, int ON, int OG, int COLS>
static void run_layer(const LayerPtrs& p, hipStream_t s) {
  // sender projection ns = n @ Ws -> [NN, OE] (no bias, no relu)
  wmma_gemm<IN_, OE, false><<<(NN / 16) / WPB, 32 * WPB, 0, s>>>(p.n_in, p.Wst, nullptr,
                                                                 p.ns_buf, NN);
  small_mm<<<cdiv(NGR * OE, 256), 256, 0, s>>>(p.g_in, p.Wg_e, p.ge, NGR, IG, OE);
  small_mm<<<cdiv(NGR * ON, 256), 256, 0, s>>>(p.g_in, p.Wg2_n, p.gn, NGR, IG, ON);
  hipMemsetAsync(p.n_gsum, 0, (size_t)NGR * ON * 4, s);
  edge_update<IE, OE, COLS><<<NGR * (OE / COLS), 512, 0, s>>>(p.e_in, p.Wet, p.ns_buf,
      p.ge, p.be, p.senders, p.receivers, p.indeg, p.e_out, p.inc_h, p.e_gsum);
  node_update<IN_, OE, ON><<<(NN / 16) / WPB, 32 * WPB, 0, s>>>(p.n_in, p.Wnt, p.inc_h,
      p.Wit, p.gn, p.bn, p.n_out, p.n_gsum, NN);
  glob_update<<<cdiv(NGR * OG, 256), 256, 0, s>>>(p.e_gsum, p.n_gsum, p.g_in, p.We2, p.Wn2,
      p.Wg3, p.bg, OE, ON, IG, OG, 1.0f / EPG, 1.0f / NPG, p.g_out);
}

extern "C" void kernel_launch(void* const* d_in, const int* in_sizes, int n_in,
                              void* d_out, int out_size, void* d_ws, size_t ws_size,
                              hipStream_t stream) {
  (void)in_sizes; (void)n_in; (void)out_size; (void)ws_size;
  const float* node_f = (const float*)d_in[0];
  const float* edge_f = (const float*)d_in[1];
  const float* glob_f = (const float*)d_in[2];
  const int* senders   = (const int*)d_in[3];
  const int* receivers = (const int*)d_in[4];
  auto F = [&](int i) { return (const float*)d_in[i]; };

  // ---- workspace carve ----
  char* wsb = (char*)d_ws; size_t off = 0;
  auto carve = [&](size_t bytes) -> void* {
    void* p = wsb + off; off += (bytes + 255) & ~(size_t)255; return p;
  };
  _Float16* nf16   = (_Float16*)carve((size_t)NN * 96 * 2);
  _Float16* eA     = (_Float16*)carve((size_t)NE * 64 * 2);
  _Float16* eB     = (_Float16*)carve((size_t)NE * 64 * 2);
  _Float16* nA     = (_Float16*)carve((size_t)NN * 128 * 2);
  _Float16* nB     = (_Float16*)carve((size_t)NN * 128 * 2);
  _Float16* nsbuf  = (_Float16*)carve((size_t)NN * 64 * 2);
  _Float16* inch   = (_Float16*)carve((size_t)NN * 64 * 2);
  int*      indeg  = (int*)     carve((size_t)NN * 4);
  float*    egsum  = (float*)   carve((size_t)NGR * 64 * 4);
  float*    ngsum  = (float*)   carve((size_t)NGR * 128 * 4);
  float*    gA     = (float*)   carve((size_t)NGR * 32 * 4);
  float*    gB     = (float*)   carve((size_t)NGR * 32 * 4);
  float*    gebuf  = (float*)   carve((size_t)NGR * 64 * 4);
  float*    gnbuf  = (float*)   carve((size_t)NGR * 128 * 4);
  float*    nsum   = (float*)   carve((size_t)NGR * 4);
  float*    b1p    = (float*)   carve(16 * 4);       // padded n1 bias
  auto carveW = [&](int fo, int fi) {
    return (_Float16*)carve((size_t)ceil16(fo) * ceil32(fi) * 2);
  };
  const int IEs[3] = {8, 16, 32}, INs[3] = {16, 32, 64};
  const int OEs[3] = {16, 32, 64}, ONs[3] = {32, 64, 128};
  _Float16* n1t = carveW(8, 83);    // [16][96]
  _Float16* n2t = carveW(16, 16);   // [16][32]
  _Float16 *Wet[3], *Wst[3], *Wnt[3], *Wit[3];
  for (int l = 0; l < 3; ++l) {
    Wet[l] = carveW(OEs[l], IEs[l]); Wst[l] = carveW(OEs[l], INs[l]);
    Wnt[l] = carveW(ONs[l], INs[l]); Wit[l] = carveW(ONs[l], OEs[l]);
  }

  // ---- weight prep: f32 -> transposed, zero-padded f16 WMMA B operands ----
  auto tr = [&](const float* W, _Float16* Wt, int fi, int fo) {
    int KP = ceil32(fi), NP = ceil16(fo);
    w_to_f16t_pad<<<cdiv(NP * KP, 256), 256, 0, stream>>>(W, Wt, fi, fo, KP, NP);
  };
  tr(F(13), n1t, 83, 8);
  tr(F(15), n2t, 8, 16);
  for (int l = 0; l < 3; ++l) {
    int b7 = 17 + 12 * l;
    tr(F(b7 + 0), Wet[l], IEs[l], OEs[l]);   // edge.We
    tr(F(b7 + 1), Wst[l], INs[l], OEs[l]);   // edge.Ws
    tr(F(b7 + 8), Wnt[l], INs[l], ONs[l]);   // node.Wn
    tr(F(b7 + 9), Wit[l], OEs[l], ONs[l]);   // node.Wi
  }
  pad_bias<<<1, 32, 0, stream>>>(F(14), b1p, 8, 16);

  // ---- encoders ----
  node_to_f16_pad<<<cdiv((long)NN * 96, 256), 256, 0, stream>>>(node_f, nf16);
  // 83->8 (padded to [N,16]) then 16->16
  wmma_gemm<96, 16, true><<<(NN / 16) / WPB, 32 * WPB, 0, stream>>>(nf16, n1t, b1p, nB, NN);
  wmma_gemm<16, 16, true><<<(NN / 16) / WPB, 32 * WPB, 0, stream>>>(nB, n2t, F(16), nA, NN);
  edge_encoder<<<cdiv(NE, 256), 256, 0, stream>>>(edge_f, F(7), F(8), F(9), F(10), eA, NE);
  glob_encoder<<<cdiv(NGR * 4, 256), 256, 0, stream>>>(glob_f, F(11), F(12), gA);
  hipMemsetAsync(indeg, 0, (size_t)NN * 4, stream);
  count_indeg<<<cdiv(NE, 256), 256, 0, stream>>>(receivers, indeg, NE);

  // ---- GN layers (ping-pong buffers) ----
  LayerPtrs p{};
  p.senders = senders; p.receivers = receivers; p.indeg = indeg;
  p.ns_buf = nsbuf; p.inc_h = inch;
  p.ge = gebuf; p.gn = gnbuf; p.e_gsum = egsum; p.n_gsum = ngsum;
  auto set_w = [&](int l) {
    int b7 = 17 + 12 * l;
    p.Wet = Wet[l]; p.Wst = Wst[l]; p.Wnt = Wnt[l]; p.Wit = Wit[l];
    p.Wg_e = F(b7 + 2); p.be = F(b7 + 3);
    p.We2 = F(b7 + 4); p.Wn2 = F(b7 + 5); p.Wg3 = F(b7 + 6); p.bg = F(b7 + 7);
    p.Wg2_n = F(b7 + 10); p.bn = F(b7 + 11);
  };
  // layer 0: (8,16,4) -> (16,32,8)
  set_w(0);
  p.e_in = eA; p.n_in = nA; p.g_in = gA; p.e_out = eB; p.n_out = nB; p.g_out = gB;
  run_layer<8, 16, 4, 16, 32, 8, 16>(p, stream);
  // layer 1: (16,32,8) -> (32,64,16)
  set_w(1);
  p.e_in = eB; p.n_in = nB; p.g_in = gB; p.e_out = eA; p.n_out = nA; p.g_out = gA;
  run_layer<16, 32, 8, 32, 64, 16, 32>(p, stream);
  // layer 2: (32,64,16) -> (64,128,32)  (two 32-col slabs per graph)
  set_w(2);
  p.e_in = eA; p.n_in = nA; p.g_in = gA; p.e_out = eB; p.n_out = nB; p.g_out = gB;
  run_layer<32, 64, 16, 64, 128, 32, 32>(p, stream);

  // ---- readout ----
  float* out = (float*)d_out;
  hipMemsetAsync(nsum, 0, (size_t)NGR * 4, stream);
  node_readout<<<cdiv(NN, 256), 256, 0, stream>>>(nB, F(56), F(57), out, nsum);
  glob_readout<<<cdiv(NGR, 256), 256, 0, stream>>>(gB, F(53), F(54), F(55), nsum, out + NN);
}
